// Head_17437567221892
// MI455X (gfx1250) — matrix operand order repaired and checked by reading, not compiled, not run on hardware
//
#include <hip/hip_runtime.h>
#include <hip/hip_bf16.h>
#include <math.h>

typedef __attribute__((ext_vector_type(16))) _Float16 v16h;
typedef __attribute__((ext_vector_type(8)))  _Float16 v8h;
typedef __attribute__((ext_vector_type(8)))  float    v8f;
typedef __attribute__((ext_vector_type(4)))  int      v4i;

union V16H { v16h v; v8h h[2]; };

#define BB 8
#define TT 2048
#define CC 1024
#define HH 64

#if defined(__AMDGCN__) && __has_builtin(__builtin_amdgcn_global_load_async_to_lds_b128)
#define HAVE_ASYNC 1
#endif

#define AS1 __attribute__((address_space(1)))
#define AS3 __attribute__((address_space(3)))

// 16-byte global -> LDS copy (async on gfx1250, staged fallback otherwise)
__device__ __forceinline__ void copy16(const _Float16* src, _Float16* dst) {
#if defined(HAVE_ASYNC)
  __builtin_amdgcn_global_load_async_to_lds_b128((AS1 v4i*)src, (AS3 v4i*)dst, 0, 0);
#else
  *(int4*)dst = *(const int4*)src;
#endif
}

__device__ __forceinline__ void wait_async_all() {
#if defined(HAVE_ASYNC)
#if __has_builtin(__builtin_amdgcn_s_wait_asynccnt)
  __builtin_amdgcn_s_wait_asynccnt(0);
#else
  asm volatile("s_wait_asynccnt 0x0" ::: "memory");
#endif
#endif
}

// ---------------- kernel 0: Wt[p][h][c] = (f16) W_p[c][h] ----------------
__global__ __launch_bounds__(256) void prep_weights(const float* __restrict__ Wq,
                                                    const float* __restrict__ Wk,
                                                    const float* __restrict__ Wv,
                                                    _Float16* __restrict__ Wt) {
  int idx = blockIdx.x * blockDim.x + threadIdx.x;   // 3*64*1024 total
  int c = idx & (CC - 1);
  int h = (idx >> 10) & (HH - 1);
  int p = idx >> 16;                                  // 0..2
  const float* W = (p == 0) ? Wq : (p == 1) ? Wk : Wv;
  Wt[idx] = (_Float16)W[c * HH + h];
}

// ---------------- kernel 1: projections via WMMA ----------------
// One wave computes a 16x64 output tile for one projection.
// Q,K stored row-major f16 [B*T][64]; V stored transposed f16 [B][64][T].
__global__ __launch_bounds__(128) void proj_kernel(const float* __restrict__ x,
                                                   const _Float16* __restrict__ Wt,
                                                   _Float16* __restrict__ Q,
                                                   _Float16* __restrict__ K,
                                                   _Float16* __restrict__ Vt) {
  const int lane  = threadIdx.x & 31;
  const int wslot = threadIdx.x >> 5;
  const int wid   = blockIdx.x * (blockDim.x >> 5) + wslot;
  const int ntile = (BB * TT) / 16;                 // 1024 row tiles
  const int tile  = wid % ntile;
  const int p     = wid / ntile;                    // 0=Q 1=K 2=V
  const int n     = lane & 15;
  const int hs    = lane >> 4;                      // lane-half select

  const int rowbase = tile * 16;
  const float* xrow = x + (size_t)(rowbase + n) * CC;     // A row M = lane&15
  const _Float16* wbase = Wt + (size_t)p * (HH * CC);

  v8f acc[4] = {};
  for (int c = 0; c < CC; c += 32) {
    // A fragment: convert f32 x -> f16 on the fly.
    // Lane needs K = {base..base+15}, base = hs*16.
    V16H a;
    {
      const float* pA = xrow + c + hs * 16;
      float4 f0 = *(const float4*)(pA);
      float4 f1 = *(const float4*)(pA + 4);
      float4 f2 = *(const float4*)(pA + 8);
      float4 f3 = *(const float4*)(pA + 12);
      v8h lo, hi;
      lo[0]=(_Float16)f0.x; lo[1]=(_Float16)f0.y; lo[2]=(_Float16)f0.z; lo[3]=(_Float16)f0.w;
      lo[4]=(_Float16)f1.x; lo[5]=(_Float16)f1.y; lo[6]=(_Float16)f1.z; lo[7]=(_Float16)f1.w;
      hi[0]=(_Float16)f2.x; hi[1]=(_Float16)f2.y; hi[2]=(_Float16)f2.z; hi[3]=(_Float16)f2.w;
      hi[4]=(_Float16)f3.x; hi[5]=(_Float16)f3.y; hi[6]=(_Float16)f3.z; hi[7]=(_Float16)f3.w;
      a.h[0] = lo; a.h[1] = hi;
    }
    for (int nt = 0; nt < 4; ++nt) {
      // B fragment: lane holds column n of Wt-tile, K = hs*16 + {0..15} contiguous.
      V16H bf;
      const _Float16* pB = wbase + (size_t)(nt * 16 + n) * CC + c + hs * 16;
      bf.h[0] = *(const v8h*)(pB);
      bf.h[1] = *(const v8h*)(pB + 8);
      acc[nt] = __builtin_amdgcn_wmma_f32_16x16x32_f16(false, a.v, false, bf.v,
                                                       (short)0, acc[nt], false, false);
    }
  }

  if (p < 2) {
    _Float16* O = (p == 0) ? Q : K;
    for (int nt = 0; nt < 4; ++nt)
      for (int r = 0; r < 8; ++r) {
        int row = rowbase + hs * 8 + r;                // C layout: M = r + hs*8
        O[(size_t)row * HH + nt * 16 + n] = (_Float16)acc[nt][r];
      }
  } else {
    int b  = rowbase / TT;
    int t0 = (rowbase % TT) + hs * 8;
    for (int nt = 0; nt < 4; ++nt) {
      v8h pk;
      for (int r = 0; r < 8; ++r) pk[r] = (_Float16)acc[nt][r];
      *(v8h*)(Vt + ((size_t)(b * HH + nt * 16 + n) * TT + t0)) = pk;  // 8 consecutive t
    }
  }
}

// ---------------- kernel 2: block-cooperative flash attention ----------------
// Block = 4 waves = 64 consecutive query rows of one batch. K/V 32-key tiles are
// double-buffered in LDS via async global->LDS copies shared by all 4 waves.
__global__ __launch_bounds__(128) void attn_kernel(const _Float16* __restrict__ Q,
                                                   const _Float16* __restrict__ K,
                                                   const _Float16* __restrict__ Vt,
                                                   float* __restrict__ O) {
  __shared__ alignas(16) _Float16 Klds[2][32 * 64];   // [buf][keyrow][hdim]  4KB each
  __shared__ alignas(16) _Float16 Vlds[2][64 * 32];   // [buf][hdim][keyrow]  4KB each
  __shared__ alignas(16) _Float16 Plds[4][16 * 32];   // per-wave P transpose tile

  const int tid   = threadIdx.x;
  const int lane  = tid & 31;
  const int wslot = tid >> 5;
  const int bpb   = TT / 64;                    // blocks per batch = 32
  const int b     = blockIdx.x / bpb;
  const int qb    = blockIdx.x % bpb;           // 64-row block index
  const int qbase = qb * 64 + wslot * 16;       // this wave's 16 query rows
  const int n     = lane & 15;
  const int hs    = lane >> 4;
  _Float16* P = Plds[wslot];

  // Q fragments for the two 32-wide hdim chunks
  V16H qa[2];
  {
    const _Float16* qrow = Q + (size_t)(b * TT + qbase + n) * HH;
    for (int ch = 0; ch < 2; ++ch) {
      const _Float16* pA = qrow + ch * 32 + hs * 16;
      qa[ch].h[0] = *(const v8h*)(pA);
      qa[ch].h[1] = *(const v8h*)(pA + 8);
    }
  }

  float mrow[8], lrow[8], p0[8], p1[8], corr[8];
  v8f acc[4] = {};
  for (int r = 0; r < 8; ++r) { mrow[r] = -INFINITY; lrow[r] = 0.f; }
  const float scale = 0.125f;                   // 1/sqrt(64)

  const int nkt_w = qbase / 32 + 1;             // this wave's causal key-tile span
  const int nkt_b = (qb * 64 + 48) / 32 + 1;    // block-wide span (last wave)

  // issue tile 0 into buffer 0 (each thread copies 2x16B of K + 2x16B of V)
  {
    const _Float16* Kg = K + (size_t)(b * TT) * HH;
    for (int i = 0; i < 2; ++i) {
      int c = tid + i * 128;                    // 0..255 16-byte chunks
      copy16(Kg + c * 8, &Klds[0][c * 8]);
      int row = c >> 2, off = (c & 3) * 8;
      copy16(Vt + (size_t)(b * HH + row) * TT + off, &Vlds[0][row * 32 + off]);
    }
  }

  for (int kt = 0; kt < nkt_b; ++kt) {
    const int buf = kt & 1;
    wait_async_all();                           // own async chunks of tile kt done
    __syncthreads();                            // all waves' chunks done + prev compute done

    if (kt + 1 < nkt_b) {                       // prefetch next tile into other buffer
      const int kb2 = (kt + 1) * 32;
      const _Float16* Kg = K + (size_t)(b * TT + kb2) * HH;
      for (int i = 0; i < 2; ++i) {
        int c = tid + i * 128;
        copy16(Kg + c * 8, &Klds[buf ^ 1][c * 8]);
        int row = c >> 2, off = (c & 3) * 8;
        copy16(Vt + (size_t)(b * HH + row) * TT + kb2 + off, &Vlds[buf ^ 1][row * 32 + off]);
      }
    }

    if (kt < nkt_w) {                           // wave-uniform: EXEC stays all-ones
      const int kbase = kt * 32;
      const _Float16* Kl = Klds[buf];
      const _Float16* Vl = Vlds[buf];
      // ---- scores: S = Q x K^T (two 16x16 column tiles, two hdim chunks each)
      v8f s0 = {}, s1 = {};
      for (int ch = 0; ch < 2; ++ch) {
        V16H b0, b1;
        const _Float16* q0 = Kl + (size_t)n * HH + ch * 32 + hs * 16;
        const _Float16* q1 = Kl + (size_t)(16 + n) * HH + ch * 32 + hs * 16;
        b0.h[0] = *(const v8h*)(q0); b0.h[1] = *(const v8h*)(q0 + 8);
        b1.h[0] = *(const v8h*)(q1); b1.h[1] = *(const v8h*)(q1 + 8);
        s0 = __builtin_amdgcn_wmma_f32_16x16x32_f16(false, qa[ch].v, false, b0.v,
                                                    (short)0, s0, false, false);
        s1 = __builtin_amdgcn_wmma_f32_16x16x32_f16(false, qa[ch].v, false, b1.v,
                                                    (short)0, s1, false, false);
      }
      // ---- causal mask + online softmax (row = r + hs*8; 16-lane row reduction)
      for (int r = 0; r < 8; ++r) {
        int qrow = qbase + hs * 8 + r;
        float v0 = (kbase + n      <= qrow) ? s0[r] * scale : -INFINITY;
        float v1 = (kbase + 16 + n <= qrow) ? s1[r] * scale : -INFINITY;
        float t = fmaxf(v0, v1);
        for (int m = 1; m <= 8; m <<= 1) t = fmaxf(t, __shfl_xor(t, m, 32));
        float mnew = fmaxf(mrow[r], t);
        float e0 = __expf(v0 - mnew);
        float e1 = __expf(v1 - mnew);
        float ss = e0 + e1;
        for (int m = 1; m <= 8; m <<= 1) ss += __shfl_xor(ss, m, 32);
        corr[r] = __expf(mrow[r] - mnew);
        lrow[r] = lrow[r] * corr[r] + ss;
        mrow[r] = mnew;
        p0[r] = e0; p1[r] = e1;
      }
      for (int nt = 0; nt < 4; ++nt)
        for (int r = 0; r < 8; ++r)
          acc[nt][r] *= corr[r];
      // ---- C-layout P -> A-layout P via per-wave LDS tile (in-order per wave)
      for (int r = 0; r < 8; ++r) {
        P[(hs * 8 + r) * 32 + n]      = (_Float16)p0[r];
        P[(hs * 8 + r) * 32 + 16 + n] = (_Float16)p1[r];
      }
      V16H pa;
      {
        const _Float16* pp = P + n * 32 + hs * 16;
        pa.h[0] = *(const v8h*)(pp);
        pa.h[1] = *(const v8h*)(pp + 8);
      }
      // ---- O += P x V  (B fragments contiguous from transposed V tile in LDS)
      for (int nt = 0; nt < 4; ++nt) {
        V16H vb;
        const _Float16* pv = Vl + (size_t)(nt * 16 + n) * 32 + hs * 16;
        vb.h[0] = *(const v8h*)(pv);
        vb.h[1] = *(const v8h*)(pv + 8);
        acc[nt] = __builtin_amdgcn_wmma_f32_16x16x32_f16(false, pa.v, false, vb.v,
                                                         (short)0, acc[nt], false, false);
      }
    }
  }

  // ---- epilogue: divide by row sums, store f32
  for (int r = 0; r < 8; ++r) lrow[r] = 1.0f / lrow[r];
  for (int nt = 0; nt < 4; ++nt)
    for (int r = 0; r < 8; ++r) {
      int row = qbase + hs * 8 + r;
      O[(size_t)(b * TT + row) * HH + nt * 16 + n] = acc[nt][r] * lrow[r];
    }
}

extern "C" void kernel_launch(void* const* d_in, const int* in_sizes, int n_in,
                              void* d_out, int out_size, void* d_ws, size_t ws_size,
                              hipStream_t stream) {
  const float* x  = (const float*)d_in[0];
  const float* Wq = (const float*)d_in[1];
  const float* Wk = (const float*)d_in[2];
  const float* Wv = (const float*)d_in[3];

  char* ws = (char*)d_ws;
  _Float16* Wt = (_Float16*)(ws);                 // 3*64*1024*2   = 393216 B
  _Float16* Q  = (_Float16*)(ws + 393216);        // 8*2048*64*2   = 2 MB
  _Float16* K  = (_Float16*)(ws + 2490368);       // 2 MB
  _Float16* Vt = (_Float16*)(ws + 4587520);       // 2 MB (total ~6.4 MB)
  float* O = (float*)d_out;

  prep_weights<<<768, 256, 0, stream>>>(Wq, Wk, Wv, Wt);          // 3*64*1024 elems
  proj_kernel <<<768, 128, 0, stream>>>(x, Wt, Q, K, Vt);         // 3072 waves
  attn_kernel <<<256, 128, 0, stream>>>(Q, K, Vt, O);             // 256 blocks x 64 rows
}
